// AttentionLayer_22445499089014
// MI455X (gfx1250) — matrix operand order repaired and checked by reading, not compiled, not run on hardware
//
#include <hip/hip_runtime.h>

#define NNODES 200000
#define NEDGES 500000
#define DMODEL 128
#define NHEADS 2
#define QSCALE 0.125f   // (D/H)^-0.5 = 64^-0.5

typedef __attribute__((ext_vector_type(16))) __bf16 v16bf;
typedef __attribute__((ext_vector_type(8)))  float  v8f;

// ---------------------------------------------------------------------------
// Order-preserving float -> uint key for atomicMax-based segment max.
// key(0) == "minus infinity" sentinel (maps to -NaN, never produced by data).
__device__ __forceinline__ unsigned fkey(float f) {
    unsigned u = __float_as_uint(f);
    return (u & 0x80000000u) ? ~u : (u | 0x80000000u);
}
__device__ __forceinline__ float funkey(unsigned k) {
    unsigned u = (k & 0x80000000u) ? (k & 0x7FFFFFFFu) : ~k;
    return __uint_as_float(u);
}

// ---------------------------------------------------------------------------
__global__ void zero_f32(float* __restrict__ p, long long n) {
    long long i = (long long)blockIdx.x * blockDim.x + threadIdx.x;
    if (i < n) p[i] = 0.0f;
}

// cw[m] = Wn @ We_r   (m = rel*3 + {K,Q,V}), 6 matrices of 128x128
__global__ void fuse_weights(const float* __restrict__ Wk, const float* __restrict__ Wq,
                             const float* __restrict__ Wv, const float* __restrict__ We0,
                             const float* __restrict__ We1, float* __restrict__ cw) {
    int i = blockIdx.x * blockDim.x + threadIdx.x;
    if (i >= 6 * DMODEL * DMODEL) return;
    int m = i >> 14;            // matrix index 0..5
    int r = (i >> 7) & 127;     // row
    int c = i & 127;            // col
    const float* Wn = (m % 3 == 0) ? Wk : ((m % 3 == 1) ? Wq : Wv);
    const float* We = (m < 3) ? We0 : We1;
    float acc = 0.0f;
    #pragma unroll 8
    for (int t = 0; t < DMODEL; ++t) acc += Wn[r * DMODEL + t] * We[t * DMODEL + c];
    cw[i] = acc;
}

// cb[m] = bn @ We_r + be_r
__global__ void fuse_bias(const float* __restrict__ bk, const float* __restrict__ bq,
                          const float* __restrict__ bv, const float* __restrict__ We0,
                          const float* __restrict__ be0, const float* __restrict__ We1,
                          const float* __restrict__ be1, float* __restrict__ cb) {
    int i = blockIdx.x * blockDim.x + threadIdx.x;
    if (i >= 6 * DMODEL) return;
    int m = i >> 7;
    int c = i & 127;
    const float* bn = (m % 3 == 0) ? bk : ((m % 3 == 1) ? bq : bv);
    const float* We = (m < 3) ? We0 : We1;
    const float* be = (m < 3) ? be0 : be1;
    float acc = be[c];
    #pragma unroll 8
    for (int t = 0; t < DMODEL; ++t) acc += bn[t] * We[t * DMODEL + c];
    cb[i] = acc;
}

// ---------------------------------------------------------------------------
// C[nrows x 128] = (A[nrows x 128] @ W[128 x 128] + bias) * scale
// bf16 WMMA, f32 accumulate. W staged in LDS pre-swizzled into B-fragment
// order so each lane's 16-element fragment is one contiguous 32-byte read.
__global__ __launch_bounds__(256)
void gemm128_bias_wmma(const float* __restrict__ A, const float* __restrict__ W,
                       const float* __restrict__ bias, float* __restrict__ C,
                       int nrows, float scale) {
    // frag f = nt*4+kb (nt: 16-col tile, kb: 32-K block); within a frag,
    // lane l's elements e=0..15 map to W[k][col] with
    //   k   = kb*32 + 16*(l>>4) + e
    //   col = nt*16 + (l&15)
    __shared__ __bf16 Wl[32 * 32 * 16];   // 32 frags x 32 lanes x 16 bf16 = 32 KB
    for (int i = threadIdx.x; i < 32 * 32 * 16; i += 256) {
        int e   = i & 15;
        int ln  = (i >> 4) & 31;
        int kb  = (i >> 9) & 3;
        int nt  = i >> 11;
        int k   = kb * 32 + 16 * (ln >> 4) + e;
        int col = nt * 16 + (ln & 15);
        Wl[i] = (__bf16)W[k * DMODEL + col];
    }
    __syncthreads();

    const int lane   = threadIdx.x & 31;
    const int wave   = threadIdx.x >> 5;
    const int tile   = blockIdx.x * 8 + wave;
    const int ntiles = nrows >> 4;           // nrows is a multiple of 16 here
    if (tile >= ntiles) return;

    const int row0 = tile << 4;
    const int half = lane >> 4;              // 0 or 1
    const int mrow = lane & 15;              // A row within tile
    const int ncol = lane & 15;              // B/D column within tile

    // --- A fragments: 16x32 bf16 per K-block, 4 blocks cover K=0..127 ---
    // layout: element e -> K = kb*32 + half*16 + e
    // i.e. one run of 16 consecutive floats: [base..base+15]
    v16bf afrag[4];
    #pragma unroll
    for (int kb = 0; kb < 4; ++kb) {
        const float* arow = A + (size_t)(row0 + mrow) * DMODEL + kb * 32 + half * 16;
        float4 t0 = *(const float4*)(arow + 0);
        float4 t1 = *(const float4*)(arow + 4);
        float4 t2 = *(const float4*)(arow + 8);
        float4 t3 = *(const float4*)(arow + 12);
        afrag[kb][0]  = (__bf16)t0.x;  afrag[kb][1]  = (__bf16)t0.y;
        afrag[kb][2]  = (__bf16)t0.z;  afrag[kb][3]  = (__bf16)t0.w;
        afrag[kb][4]  = (__bf16)t1.x;  afrag[kb][5]  = (__bf16)t1.y;
        afrag[kb][6]  = (__bf16)t1.z;  afrag[kb][7]  = (__bf16)t1.w;
        afrag[kb][8]  = (__bf16)t2.x;  afrag[kb][9]  = (__bf16)t2.y;
        afrag[kb][10] = (__bf16)t2.z;  afrag[kb][11] = (__bf16)t2.w;
        afrag[kb][12] = (__bf16)t3.x;  afrag[kb][13] = (__bf16)t3.y;
        afrag[kb][14] = (__bf16)t3.z;  afrag[kb][15] = (__bf16)t3.w;
    }

    // --- loop over 8 column tiles of 16 ---
    #pragma unroll
    for (int nt = 0; nt < 8; ++nt) {
        const int col0 = nt * 16;
        v8f acc = {};
        #pragma unroll
        for (int kb = 0; kb < 4; ++kb) {
            v16bf bfrag = *(const v16bf*)&Wl[(((nt * 4 + kb) * 32) + lane) * 16];
            acc = __builtin_amdgcn_wmma_f32_16x16x32_bf16(
                false, afrag[kb], false, bfrag, (short)0, acc, false, false);
        }
        const float b = bias[col0 + ncol];
        // D layout: VGPR i -> row m = i + 8*half, col = ncol
        #pragma unroll
        for (int i = 0; i < 8; ++i) {
            int m = i + 8 * half;
            C[(size_t)(row0 + m) * DMODEL + col0 + ncol] = (acc[i] + b) * scale;
        }
    }
}

// ---------------------------------------------------------------------------
// One wave per edge: logits[e,h] = dot64(K[src,h,:], Q[dst,h,:]); atomicMax key.
__global__ __launch_bounds__(256)
void edge_logits(const float* __restrict__ K, const float* __restrict__ Q,
                 const int* __restrict__ src, const int* __restrict__ dst,
                 float* __restrict__ logits, unsigned* __restrict__ mkeys,
                 int nedges) {
    int e = blockIdx.x * 8 + (threadIdx.x >> 5);
    int lane = threadIdx.x & 31;
    if (e >= nedges) return;
    int s = src[e], d = dst[e];
    const float2* kp = (const float2*)(K + (size_t)s * DMODEL);
    const float2* qp = (const float2*)(Q + (size_t)d * DMODEL);
    float2 k0 = kp[lane],      q0 = qp[lane];        // head 0: dims 2*lane,2*lane+1
    float2 k1 = kp[32 + lane], q1 = qp[32 + lane];   // head 1
    float acc0 = k0.x * q0.x + k0.y * q0.y;
    float acc1 = k1.x * q1.x + k1.y * q1.y;
    #pragma unroll
    for (int off = 16; off; off >>= 1) {
        acc0 += __shfl_xor(acc0, off, 32);
        acc1 += __shfl_xor(acc1, off, 32);
    }
    if (lane == 0) {
        logits[(size_t)e * 2 + 0] = acc0;
        logits[(size_t)e * 2 + 1] = acc1;
        atomicMax(&mkeys[d * 2 + 0], fkey(acc0));
        atomicMax(&mkeys[d * 2 + 1], fkey(acc1));
    }
}

// exp(logit - m[dst]) in place; segment sum via atomicAdd.
__global__ void edge_exp(const int* __restrict__ dst, float* __restrict__ logits,
                         const unsigned* __restrict__ mkeys, float* __restrict__ ssum,
                         int nedges) {
    int i = blockIdx.x * blockDim.x + threadIdx.x;
    if (i >= nedges * NHEADS) return;
    int e = i >> 1, h = i & 1;
    int d = dst[e];
    float m = funkey(mkeys[d * 2 + h]);
    float v = __expf(logits[i] - m);
    logits[i] = v;
    atomicAdd(&ssum[d * 2 + h], v);
}

// upd[dst, g*4..g*4+3] += V[src, ...] * attn(e, head); thread per (edge, float4).
__global__ void scatter_msg(const int* __restrict__ src, const int* __restrict__ dst,
                            const float* __restrict__ V, const float* __restrict__ ex,
                            const float* __restrict__ ssum, float* __restrict__ upd,
                            int nedges) {
    long long i = (long long)blockIdx.x * blockDim.x + threadIdx.x;
    if (i >= (long long)nedges * 32) return;
    int e = (int)(i >> 5);
    int g = (int)(i & 31);        // float4 group: dims g*4..g*4+3
    int h = g >> 4;               // head
    int s = src[e], d = dst[e];
    float a = ex[(size_t)e * 2 + h] / ssum[d * 2 + h];
    float4 v = *(const float4*)(V + (size_t)s * DMODEL + g * 4);
    float* u = upd + (size_t)d * DMODEL + g * 4;
    atomicAdd(u + 0, v.x * a);
    atomicAdd(u + 1, v.y * a);
    atomicAdd(u + 2, v.z * a);
    atomicAdd(u + 3, v.w * a);
}

// ---------------------------------------------------------------------------
extern "C" void kernel_launch(void* const* d_in, const int* in_sizes, int n_in,
                              void* d_out, int out_size, void* d_ws, size_t ws_size,
                              hipStream_t stream) {
    const float* feat = (const float*)d_in[0];
    const int*   src0 = (const int*)d_in[1];
    const int*   dst0 = (const int*)d_in[2];
    const int*   src1 = (const int*)d_in[3];
    const int*   dst1 = (const int*)d_in[4];
    const float* Wk = (const float*)d_in[5];
    const float* bk = (const float*)d_in[6];
    const float* Wq = (const float*)d_in[7];
    const float* bq = (const float*)d_in[8];
    const float* Wv = (const float*)d_in[9];
    const float* bv = (const float*)d_in[10];
    const float* Wo = (const float*)d_in[11];
    const float* bo = (const float*)d_in[12];
    const float* We0 = (const float*)d_in[13];
    const float* be0 = (const float*)d_in[14];
    const float* We1 = (const float*)d_in[15];
    const float* be1 = (const float*)d_in[16];
    float* out = (float*)d_out;

    const size_t ND = (size_t)NNODES * DMODEL;
    const size_t EH = (size_t)NEDGES * NHEADS;
    const size_t NH = (size_t)NNODES * NHEADS;

    float*    Kb     = (float*)d_ws;
    float*    Qb     = Kb + ND;
    float*    Vb     = Qb + ND;
    float*    upd    = Vb + ND;
    float*    logits = upd + ND;
    float*    ssum   = logits + EH;
    unsigned* mkeys  = (unsigned*)(ssum + NH);
    float*    cw     = (float*)(mkeys + NH);
    float*    cb     = cw + 6 * DMODEL * DMODEL;
    size_t needed = (size_t)((char*)(cb + 6 * DMODEL) - (char*)d_ws);
    if (ws_size < needed) return;

    fuse_weights<<<(6 * DMODEL * DMODEL + 255) / 256, 256, 0, stream>>>(Wk, Wq, Wv, We0, We1, cw);
    fuse_bias<<<(6 * DMODEL + 255) / 256, 256, 0, stream>>>(bk, bq, bv, We0, be0, We1, be1, cb);
    zero_f32<<<(int)((ND + 255) / 256), 256, 0, stream>>>(upd, (long long)ND);

    const int gemm_blocks = ((NNODES / 16) + 7) / 8;   // 8 waves (16-row tiles) per block
    for (int rel = 0; rel < 2; ++rel) {
        const int* src = rel ? src1 : src0;
        const int* dst = rel ? dst1 : dst0;
        // zero ssum + mkeys together (contiguous, key 0 == -inf sentinel)
        zero_f32<<<(int)((2 * NH + 255) / 256), 256, 0, stream>>>(ssum, (long long)(2 * NH));

        const float* cwK = cw + (size_t)(rel * 3 + 0) * DMODEL * DMODEL;
        const float* cwQ = cw + (size_t)(rel * 3 + 1) * DMODEL * DMODEL;
        const float* cwV = cw + (size_t)(rel * 3 + 2) * DMODEL * DMODEL;
        const float* cbK = cb + (rel * 3 + 0) * DMODEL;
        const float* cbQ = cb + (rel * 3 + 1) * DMODEL;
        const float* cbV = cb + (rel * 3 + 2) * DMODEL;

        gemm128_bias_wmma<<<gemm_blocks, 256, 0, stream>>>(feat, cwK, cbK, Kb, NNODES, 1.0f);
        gemm128_bias_wmma<<<gemm_blocks, 256, 0, stream>>>(feat, cwQ, cbQ, Qb, NNODES, QSCALE);
        gemm128_bias_wmma<<<gemm_blocks, 256, 0, stream>>>(feat, cwV, cbV, Vb, NNODES, 1.0f);

        edge_logits<<<(NEDGES + 7) / 8, 256, 0, stream>>>(Kb, Qb, src, dst, logits, mkeys, NEDGES);
        edge_exp<<<(NEDGES * NHEADS + 255) / 256, 256, 0, stream>>>(dst, logits, mkeys, ssum, NEDGES);
        scatter_msg<<<(int)(((long long)NEDGES * 32 + 255) / 256), 256, 0, stream>>>(
            src, dst, Vb, logits, ssum, upd, NEDGES);
    }

    // out = upd @ Wo + bo
    gemm128_bias_wmma<<<gemm_blocks, 256, 0, stream>>>(upd, Wo, bo, out, NNODES, 1.0f);
}
